// MultiHeadAttention_29695403885287
// MI455X (gfx1250) — compile-verified, matrix operand order checked by reading, not run on hardware
//
#include <hip/hip_runtime.h>
#include <hip/hip_bf16.h>
#include <stdint.h>

// ---------------------------------------------------------------------------
// MHA forward on gfx1250: bf16 WMMA everywhere, async-to-LDS staged B tiles,
// single raw split-barrier per k-step, fully-unrolled pipeline,
// 32x64 per-wave GEMM tiles. B=2, S=2048, D=768, H=12, HD=64.
// ---------------------------------------------------------------------------

#define B_  2
#define S_  2048
#define D_  768
#define H_  12
#define HD_ 64
#define M_  (B_ * S_)          // 4096 rows of x
#define KSTEPS_ (D_ / 32)      // 24

typedef __attribute__((ext_vector_type(16))) __bf16 v16bf;
typedef __attribute__((ext_vector_type(8)))  float  v8f;

union Frag {
    v16bf v;
    uint4 u[2];
    unsigned short h[16];
};

__device__ __forceinline__ unsigned short f2bf(float f) {
    unsigned int u = __float_as_uint(f);
    unsigned int r = u + 0x7FFFu + ((u >> 16) & 1u);   // round-to-nearest-even
    return (unsigned short)(r >> 16);
}

// Load a 16x32 bf16 fragment (A-layout; also used for B via transposed operand).
// Lane l: row = l&15, kb = 8*(l>>4); elems 0..7 = K kb..kb+7, 8..15 = kb+16..kb+23.
__device__ __forceinline__ v16bf load_frag(const unsigned short* __restrict__ base, int ld) {
    int lane = threadIdx.x & 31;
    int row  = lane & 15;
    int kb   = (lane >> 4) * 8;
    Frag f;
    const unsigned short* p = base + (size_t)row * ld + kb;
    f.u[0] = *(const uint4*)(p);
    f.u[1] = *(const uint4*)(p + 16);
    return f.v;
}

__device__ __forceinline__ v8f wmma_bf16(v16bf a, v16bf b, v8f c) {
    return __builtin_amdgcn_wmma_f32_16x16x32_bf16(false, a, false, b, (short)0, c, false, false);
}

// In-wave LDS fence: DS pipe is in-order per wave; just drain DScnt.
__device__ __forceinline__ void wave_lds_fence() {
    __builtin_amdgcn_wave_barrier();
    asm volatile("s_wait_dscnt 0" ::: "memory");
    __builtin_amdgcn_wave_barrier();
}

// Raw workgroup split barrier WITHOUT the vmem-drain fence of __syncthreads().
// Async-copy completion is handled explicitly via s_wait_asynccnt; ds reads are
// always drained before the WMMAs that precede the barrier.
__device__ __forceinline__ void block_barrier() {
    asm volatile("s_barrier_signal -1\n\ts_barrier_wait -1" ::: "memory");
}

// Async global->LDS copy of 16 bytes per lane (gfx1250, tracked by ASYNCcnt).
__device__ __forceinline__ void async_copy_b128(unsigned int lds_addr, const void* gptr) {
    unsigned long long g = (unsigned long long)(uintptr_t)gptr;
    asm volatile("global_load_async_to_lds_b128 %0, %1, off"
                 :: "v"(lds_addr), "v"(g)
                 : "memory");
}
__device__ __forceinline__ void wait_async0() { asm volatile("s_wait_asynccnt 0x0" ::: "memory"); }

// ---------------------------------------------------------------------------
// Kernel 1: fp32 -> bf16 convert; weights transposed to [N,K] bf16.
// ---------------------------------------------------------------------------
__global__ void convert_kernel(const float* __restrict__ x,
                               const float* __restrict__ wq,
                               const float* __restrict__ wk,
                               const float* __restrict__ wv,
                               const float* __restrict__ wo,
                               unsigned short* __restrict__ xbf,
                               unsigned short* __restrict__ wqt,
                               unsigned short* __restrict__ wkt,
                               unsigned short* __restrict__ wvt,
                               unsigned short* __restrict__ wot) {
    int idx = blockIdx.x * blockDim.x + threadIdx.x;
    if (idx < M_ * D_) xbf[idx] = f2bf(x[idx]);
    if (idx < D_ * D_) {
        int n = idx / D_;
        int k = idx - n * D_;
        int src = k * D_ + n;            // transpose: [K,N] -> [N,K]
        wqt[idx] = f2bf(wq[src]);
        wkt[idx] = f2bf(wk[src]);
        wvt[idx] = f2bf(wv[src]);
        wot[idx] = f2bf(wo[src]);
    }
}

// ---------------------------------------------------------------------------
// Block GEMM core: 8 waves, block tile 256(M) x 64(N); each wave owns a
// 32(M) x 64(N) tile (8 accumulators). K = 768 in steps of 32. The 64x32 B
// tile is staged to LDS with one async b128 per lane, double buffered, ONE
// raw barrier per k-step: wait stage(i) -> barrier -> issue stage(i+1) ->
// compute buf(i). A fragments register-double-buffered. Fully unrolled.
// ---------------------------------------------------------------------------
__device__ __forceinline__ void gemm_block_core(const unsigned short* __restrict__ Awave, // &A[m0_wave, 0], ld = D_
                                                const unsigned short* __restrict__ Wt,    // [D_, D_] (N,K)
                                                int n0,
                                                unsigned short* __restrict__ sB,          // shared: 2 * 64 * 32
                                                v8f acc[8]) {
    const int tid  = threadIdx.x;
    const int brow = tid >> 2;            // 0..63  (row within B tile)
    const int bck  = (tid & 3) * 8;       // element offset within 32-wide K slab

    const unsigned short* gW = Wt + (size_t)(n0 + brow) * D_ + bck;
    unsigned int lB = (unsigned int)(uintptr_t)(sB + brow * 32 + bck);

    async_copy_b128(lB, gW);                               // stage 0
    v16bf a0 = load_frag(Awave, D_);
    v16bf a1 = load_frag(Awave + (size_t)16 * D_, D_);

#pragma unroll
    for (int i = 0; i < KSTEPS_; ++i) {
        const bool more = (i + 1) < KSTEPS_;

        wait_async0();                                     // stage i complete (issued 1 iter ago)
        block_barrier();                                   // all stages i done; all buf(i-1) readers done
        if (more)
            async_copy_b128(lB + (((i + 1) & 1) * (64 * 32 * 2)), gW + (i + 1) * 32);

        v16bf a0n = a0, a1n = a1;
        if (more) {
            a0n = load_frag(Awave + (i + 1) * 32, D_);
            a1n = load_frag(Awave + (size_t)16 * D_ + (i + 1) * 32, D_);
        }

        const unsigned short* bb = sB + (i & 1) * (64 * 32);
        v16bf b0 = load_frag(bb + 0 * 16 * 32, 32);
        v16bf b1 = load_frag(bb + 1 * 16 * 32, 32);
        v16bf b2 = load_frag(bb + 2 * 16 * 32, 32);
        v16bf b3 = load_frag(bb + 3 * 16 * 32, 32);

        acc[0] = wmma_bf16(a0, b0, acc[0]);
        acc[1] = wmma_bf16(a0, b1, acc[1]);
        acc[2] = wmma_bf16(a0, b2, acc[2]);
        acc[3] = wmma_bf16(a0, b3, acc[3]);
        acc[4] = wmma_bf16(a1, b0, acc[4]);
        acc[5] = wmma_bf16(a1, b1, acc[5]);
        acc[6] = wmma_bf16(a1, b2, acc[6]);
        acc[7] = wmma_bf16(a1, b3, acc[7]);

        a0 = a0n; a1 = a1n;
    }
}

// ---------------------------------------------------------------------------
// Kernel 2: QKV projection GEMMs. Block tile 256x64; blockIdx.y selects
// Q(0)/K(1)/V(2). Q gets 1/sqrt(HD) folded in. Outputs: Q,K as [B,H,S,HD];
// V transposed as [B,H,HD,S] (stored with packed b128 along S).
// ---------------------------------------------------------------------------
__global__ void __launch_bounds__(256, 1)
qkv_kernel(const unsigned short* __restrict__ xbf,
           const unsigned short* __restrict__ wqt,
           const unsigned short* __restrict__ wkt,
           const unsigned short* __restrict__ wvt,
           const float* __restrict__ bq,
           const float* __restrict__ bk,
           const float* __restrict__ bv,
           unsigned short* __restrict__ Qb,
           unsigned short* __restrict__ Kb,
           unsigned short* __restrict__ Vtb) {
    __shared__ __align__(16) unsigned short sB[2 * 64 * 32];

    const int which = blockIdx.y;
    const unsigned short* W = (which == 0) ? wqt : (which == 1) ? wkt : wvt;
    const float* bias       = (which == 0) ? bq  : (which == 1) ? bk  : bv;

    const int widx = threadIdx.x >> 5;
    const int lane = threadIdx.x & 31;
    const int mtb  = blockIdx.x / (D_ / 64);               // 0..15
    const int ntb  = blockIdx.x - mtb * (D_ / 64);         // 0..11
    const int m0   = mtb * 256 + widx * 32;
    const int n0   = ntb * 64;

    v8f acc[8] = {};
    gemm_block_core(xbf + (size_t)m0 * D_, W, n0, sB, acc);

    const int hi = lane >> 4;
    const int nn = lane & 15;
#pragma unroll
    for (int t = 0; t < 4; ++t) {
        int n = n0 + t * 16 + nn;
        float bb = bias[n];
        int h  = n / HD_;
        int hd = n - h * HD_;
#pragma unroll
        for (int rr = 0; rr < 2; ++rr) {
            if (which == 2) {
                // Vt layout: 8 accumulator rows are consecutive s -> one b128
                int mbase = m0 + rr * 16 + 8 * hi;          // multiple of 8
                int b = mbase / S_;
                int s = mbase - b * S_;
                Frag pk;
#pragma unroll
                for (int r = 0; r < 8; ++r)
                    pk.h[r] = f2bf(acc[rr * 4 + t][r] + bb);
                *(uint4*)&Vtb[(((size_t)b * H_ + h) * HD_ + hd) * S_ + s] = pk.u[0];
            } else {
#pragma unroll
                for (int r = 0; r < 8; ++r) {
                    int m = m0 + rr * 16 + r + 8 * hi;
                    int b = m / S_;
                    int s = m - b * S_;
                    float val = acc[rr * 4 + t][r] + bb;
                    if (which == 0) val *= 0.125f;          // 1/sqrt(64)
                    unsigned short o = f2bf(val);
                    if (which == 0)
                        Qb[(((size_t)b * H_ + h) * S_ + s) * HD_ + hd] = o;
                    else
                        Kb[(((size_t)b * H_ + h) * S_ + s) * HD_ + hd] = o;
                }
            }
        }
    }
}

// ---------------------------------------------------------------------------
// Kernel 3: causal flash attention. One wave per (b,h, 16-query tile).
// Key tiles of 32. V fragments prefetched before the score WMMAs. Row-sum of
// P done on the matrix pipe (P x ones) instead of lane shuffles.
// ---------------------------------------------------------------------------
__global__ void __launch_bounds__(256, 1)
attn_kernel(const unsigned short* __restrict__ Qb,
            const unsigned short* __restrict__ Kb,
            const unsigned short* __restrict__ Vtb,
            unsigned short* __restrict__ ctx) {
    __shared__ __align__(16) unsigned short plds[8][16 * 32];

    const int widx = threadIdx.x >> 5;
    const int lane = threadIdx.x & 31;
    const int gwave = blockIdx.x * 8 + widx;   // 0..3071
    const int tq = gwave & (S_ / 16 - 1);      // 0..127
    const int bh = gwave >> 7;                 // 0..23
    const int q0 = tq * 16;

    const unsigned short* Qh = Qb  + (size_t)bh * S_ * HD_;
    const unsigned short* Kh = Kb  + (size_t)bh * S_ * HD_;
    const unsigned short* Vh = Vtb + (size_t)bh * HD_ * S_;

    v16bf qa0 = load_frag(Qh + (size_t)q0 * HD_ + 0,  HD_);
    v16bf qa1 = load_frag(Qh + (size_t)q0 * HD_ + 32, HD_);

    // all-ones bf16 B fragment for matrix-pipe row sums
    Frag onesf;
#pragma unroll
    for (int i = 0; i < 16; ++i) onesf.h[i] = 0x3F80;      // 1.0bf
    const v16bf ones = onesf.v;

    v8f o[4] = {};
    float mrun[8], lrun[8];
#pragma unroll
    for (int r = 0; r < 8; ++r) { mrun[r] = -1e30f; lrun[r] = 0.f; }

    const int hi = lane >> 4;
    const int nn = lane & 15;
    const int ktiles = (q0 + 15) / 32 + 1;

    for (int kt = 0; kt < ktiles; ++kt) {
        const int k0 = kt * 32;
        // ---- issue K loads, then V loads (V stays in flight through softmax)
        v16bf kb0 = load_frag(Kh + (size_t)(k0) * HD_ + 0,  HD_);
        v16bf kb1 = load_frag(Kh + (size_t)(k0) * HD_ + 32, HD_);
        v16bf kb2 = load_frag(Kh + (size_t)(k0 + 16) * HD_ + 0,  HD_);
        v16bf kb3 = load_frag(Kh + (size_t)(k0 + 16) * HD_ + 32, HD_);
        v16bf vb0 = load_frag(Vh + (size_t)(0)  * S_ + k0, S_);
        v16bf vb1 = load_frag(Vh + (size_t)(16) * S_ + k0, S_);
        v16bf vb2 = load_frag(Vh + (size_t)(32) * S_ + k0, S_);
        v16bf vb3 = load_frag(Vh + (size_t)(48) * S_ + k0, S_);

        // ---- scores S[16 x 32] = Q(16x64) . K^T ----
        v8f s0 = {}, s1 = {};
        s0 = wmma_bf16(qa0, kb0, s0);
        s0 = wmma_bf16(qa1, kb1, s0);
        s1 = wmma_bf16(qa0, kb2, s1);
        s1 = wmma_bf16(qa1, kb3, s1);

        // ---- causal mask + row max (butterfly over the 16-lane half) ----
        float tmax[8];
#pragma unroll
        for (int r = 0; r < 8; ++r) {
            int q = q0 + r + 8 * hi;
            if (k0 + nn > q)      s0[r] = -1e30f;
            if (k0 + 16 + nn > q) s1[r] = -1e30f;
            float t = fmaxf(s0[r], s1[r]);
            t = fmaxf(t, __shfl_xor(t, 1, 32));
            t = fmaxf(t, __shfl_xor(t, 2, 32));
            t = fmaxf(t, __shfl_xor(t, 4, 32));
            t = fmaxf(t, __shfl_xor(t, 8, 32));
            tmax[r] = t;
        }
        // ---- online softmax update; stage P into per-wave LDS ----
        float alpha[8];
#pragma unroll
        for (int r = 0; r < 8; ++r) {
            float mnew = fmaxf(mrun[r], tmax[r]);
            alpha[r]   = __expf(mrun[r] - mnew);
            float p0 = __expf(s0[r] - mnew);
            float p1 = __expf(s1[r] - mnew);
            mrun[r] = mnew;
            o[0][r] *= alpha[r]; o[1][r] *= alpha[r];
            o[2][r] *= alpha[r]; o[3][r] *= alpha[r];
            int row = r + 8 * hi;
            plds[widx][row * 32 + nn]      = f2bf(p0);
            plds[widx][row * 32 + 16 + nn] = f2bf(p1);
        }
        wave_lds_fence();
        // ---- reload P as an A-fragment (16x32) ----
        v16bf pa;
        {
            int row = lane & 15;
            int kb  = (lane >> 4) * 8;
            Frag f;
            const unsigned short* p = &plds[widx][row * 32 + kb];
            f.u[0] = *(const uint4*)(p);
            f.u[1] = *(const uint4*)(p + 16);
            pa = f.v;
        }
        // ---- row sums on the matrix pipe: rs[m, *] = sum_k P[m,k] ----
        v8f rs = {};
        rs = wmma_bf16(pa, ones, rs);
#pragma unroll
        for (int r = 0; r < 8; ++r)
            lrun[r] = lrun[r] * alpha[r] + rs[r];

        // ---- ctx accumulation: O(16x64) += P(16x32) . V(32x64) ----
        o[0] = wmma_bf16(pa, vb0, o[0]);
        o[1] = wmma_bf16(pa, vb1, o[1]);
        o[2] = wmma_bf16(pa, vb2, o[2]);
        o[3] = wmma_bf16(pa, vb3, o[3]);
    }

    // ---- normalize and store ctx as [B*S, D] bf16 (head-interleaved cols) ----
    const int b = bh / H_;
    const int h = bh - b * H_;
#pragma unroll
    for (int r = 0; r < 8; ++r) {
        int s = q0 + r + 8 * hi;
        float inv = 1.0f / lrun[r];
#pragma unroll
        for (int t = 0; t < 4; ++t) {
            int col = h * HD_ + t * 16 + nn;
            ctx[((size_t)(b * S_ + s)) * D_ + col] = f2bf(o[t][r] * inv);
        }
    }
}

// ---------------------------------------------------------------------------
// Kernel 4: output projection, fp32 epilogue into d_out.
// ---------------------------------------------------------------------------
__global__ void __launch_bounds__(256, 1)
oproj_kernel(const unsigned short* __restrict__ ctx,
             const unsigned short* __restrict__ wot,
             const float* __restrict__ bo,
             float* __restrict__ out) {
    __shared__ __align__(16) unsigned short sB[2 * 64 * 32];

    const int widx = threadIdx.x >> 5;
    const int lane = threadIdx.x & 31;
    const int mtb  = blockIdx.x / (D_ / 64);
    const int ntb  = blockIdx.x - mtb * (D_ / 64);
    const int m0   = mtb * 256 + widx * 32;
    const int n0   = ntb * 64;

    v8f acc[8] = {};
    gemm_block_core(ctx + (size_t)m0 * D_, wot, n0, sB, acc);

    const int hi = lane >> 4;
    const int nn = lane & 15;
#pragma unroll
    for (int t = 0; t < 4; ++t) {
        int n = n0 + t * 16 + nn;
        float bb = bo[n];
#pragma unroll
        for (int rr = 0; rr < 2; ++rr) {
#pragma unroll
            for (int r = 0; r < 8; ++r) {
                int m = m0 + rr * 16 + r + 8 * hi;
                out[(size_t)m * D_ + n] = acc[rr * 4 + t][r] + bb;
            }
        }
    }
}

// ---------------------------------------------------------------------------
// Launch
// ---------------------------------------------------------------------------
extern "C" void kernel_launch(void* const* d_in, const int* in_sizes, int n_in,
                              void* d_out, int out_size, void* d_ws, size_t ws_size,
                              hipStream_t stream) {
    const float* x  = (const float*)d_in[0];
    const float* wq = (const float*)d_in[1];
    const float* bq = (const float*)d_in[2];
    const float* wk = (const float*)d_in[3];
    const float* bk = (const float*)d_in[4];
    const float* wv = (const float*)d_in[5];
    const float* bv = (const float*)d_in[6];
    const float* wo = (const float*)d_in[7];
    const float* bo = (const float*)d_in[8];
    float* out = (float*)d_out;

    char* ws = (char*)d_ws;
    const size_t XBF  = 0;                                    // 4096*768*2
    const size_t WQT  = XBF + (size_t)M_ * D_ * 2;
    const size_t WKT  = WQT + (size_t)D_ * D_ * 2;
    const size_t WVT  = WKT + (size_t)D_ * D_ * 2;
    const size_t WOT  = WVT + (size_t)D_ * D_ * 2;
    const size_t QB   = WOT + (size_t)D_ * D_ * 2;
    const size_t KB   = QB  + (size_t)M_ * D_ * 2;
    const size_t VTB  = KB  + (size_t)M_ * D_ * 2;
    const size_t CTX  = VTB + (size_t)M_ * D_ * 2;

    unsigned short* xbf = (unsigned short*)(ws + XBF);
    unsigned short* wqt = (unsigned short*)(ws + WQT);
    unsigned short* wkt = (unsigned short*)(ws + WKT);
    unsigned short* wvt = (unsigned short*)(ws + WVT);
    unsigned short* wot = (unsigned short*)(ws + WOT);
    unsigned short* Qb  = (unsigned short*)(ws + QB);
    unsigned short* Kb  = (unsigned short*)(ws + KB);
    unsigned short* Vtb = (unsigned short*)(ws + VTB);
    unsigned short* ctx = (unsigned short*)(ws + CTX);

    convert_kernel<<<(M_ * D_ + 255) / 256, 256, 0, stream>>>(
        x, wq, wk, wv, wo, xbf, wqt, wkt, wvt, wot);

    qkv_kernel<<<dim3(192, 3), 256, 0, stream>>>(
        xbf, wqt, wkt, wvt, bq, bk, bv, Qb, Kb, Vtb);

    attn_kernel<<<384, 256, 0, stream>>>(Qb, Kb, Vtb, ctx);

    oproj_kernel<<<192, 256, 0, stream>>>(ctx, wot, bo, out);
}
